// GATLayer_72215580115373
// MI455X (gfx1250) — matrix operand order repaired and checked by reading, not compiled, not run on hardware
//
#include <hip/hip_runtime.h>

#define BC 4
#define NODES 4096
#define FEAT 64
#define OUT_DIM 32
#define MWORDS 128            // NODES/32 mask words per row
#define WAVES 4
#define ROWS_PER_WG (WAVES * 16)

typedef __attribute__((ext_vector_type(16))) _Float16 v16h;
typedef __attribute__((ext_vector_type(8)))  float    v8f;

union V16H { v16h h; unsigned int u[8]; };

// ---------------- mask build ----------------
__global__ void zero_mask_kernel(unsigned int* __restrict__ m, int nwords) {
    int i = blockIdx.x * blockDim.x + threadIdx.x;
    if (i < nwords) m[i] = 0u;
}

__global__ void build_mask_kernel(const int* __restrict__ row,
                                  const int* __restrict__ col,
                                  unsigned int* __restrict__ m, int nedges) {
    int i = blockIdx.x * blockDim.x + threadIdx.x;
    if (i < nedges) {
        int r = row[i], c = col[i];
        atomicOr(&m[r * MWORDS + (c >> 5)], 1u << (c & 31));
    }
}

// ---------------- z = h @ W (f32 -> f16) ----------------
__global__ void compute_z_kernel(const float* __restrict__ h,
                                 const float* __restrict__ W,
                                 _Float16* __restrict__ z) {
    int idx = blockIdx.x * blockDim.x + threadIdx.x;   // BC*NODES*OUT_DIM
    int d = idx & (OUT_DIM - 1);
    int n = idx >> 5;                                  // flattened (b, node)
    const float* hr = h + (size_t)n * FEAT;
    float acc = 0.f;
#pragma unroll 16
    for (int f = 0; f < FEAT; ++f)
        acc = fmaf(hr[f], W[f * OUT_DIM + d], acc);
    z[idx] = (_Float16)acc;
}

// ---------------- fused masked-attention + aggregate ----------------
__global__ __launch_bounds__(WAVES * 32)
void gat_attn_kernel(const _Float16* __restrict__ zh,
                     const unsigned int* __restrict__ mask,
                     float* __restrict__ out) {
    __shared__ __align__(16) _Float16 zkey [32 * 32];  // [key][dim]
    __shared__ __align__(16) _Float16 zkeyT[32 * 32];  // [dim][key]
    __shared__ __align__(16) _Float16 psm[WAVES][16 * 32]; // per-wave P tile [M][K]

    const int tid  = threadIdx.x;
    const int wave = tid >> 5;
    const int lane = tid & 31;
    const int l16  = lane & 15;
    const int lh   = lane >> 4;       // lane half: 0 or 1

    const int blocksPerBatch = NODES / ROWS_PER_WG;
    const int b       = blockIdx.x / blocksPerBatch;
    const int rowbase = (blockIdx.x % blocksPerBatch) * ROWS_PER_WG + wave * 16;

    const _Float16* zb = zh + (size_t)b * NODES * OUT_DIM;

    // ---- query A fragment (16x32 f16), held for the whole key sweep ----
    V16H qa;
    {
        const _Float16* qrow = zb + (size_t)(rowbase + l16) * OUT_DIM;
#pragma unroll
        for (int v = 0; v < 8; ++v) {
            int Kp = 16 * lh + 2 * v;
            qa.u[v] = *(const unsigned int*)(qrow + Kp);
        }
    }

    const unsigned int* mrow = mask + (size_t)(rowbase + l16) * MWORDS;

    float mrun[8], lrun[8];
    v8f O0 = {}; v8f O1 = {};
#pragma unroll
    for (int r = 0; r < 8; ++r) { mrun[r] = -INFINITY; lrun[r] = 0.f; }

    _Float16* ps = psm[wave];

    for (int j = 0; j < NODES; j += 32) {
        if (j + 32 < NODES) {
            // pull next 2KB key tile toward the caches (global_prefetch_b8)
            const _Float16* nf = zb + (size_t)(j + 32) * OUT_DIM;
            __builtin_prefetch(nf + (size_t)(tid & 15) * 64, 0, 1);
        }
        __syncthreads();   // everyone done reading previous tile
        {
            // cooperative stage: 128 threads x 16B = 2KB tile, both layouts
            int krow = tid >> 2;          // 0..31 key within tile
            int kcol = (tid & 3) * 8;     // 0,8,16,24 dim start
            uint4 d = *(const uint4*)(zb + (size_t)(j + krow) * OUT_DIM + kcol);
            *(uint4*)&zkey[krow * 32 + kcol] = d;
            const _Float16* t = (const _Float16*)&d;
#pragma unroll
            for (int i = 0; i < 8; ++i)
                zkeyT[(kcol + i) * 32 + krow] = t[i];
        }
        __syncthreads();

        // ---- score WMMAs: S = Zq (16x32) x Zkᵀ  (two 16-key column tiles) ----
        V16H kb0, kb1;
#pragma unroll
        for (int v = 0; v < 8; ++v) {
            kb0.u[v] = *(const unsigned int*)&zkey[ l16       * 32 + 16 * lh + 2 * v];
            kb1.u[v] = *(const unsigned int*)&zkey[(16 + l16) * 32 + 16 * lh + 2 * v];
        }
        v8f Z = {};
        v8f S0 = __builtin_amdgcn_wmma_f32_16x16x32_f16(false, qa.h, false, kb0.h,
                                                        (short)0, Z, false, false);
        v8f S1 = __builtin_amdgcn_wmma_f32_16x16x32_f16(false, qa.h, false, kb1.h,
                                                        (short)0, Z, false, false);

        unsigned int w = mrow[j >> 5];    // 32 mask bits for row (rowbase+l16)

        float p0[8], p1[8];
#pragma unroll
        for (int r = 0; r < 8; ++r) {
            // mask word for this fragment row M = r + 8*lh lives in lane (r + 8*lh)
            unsigned int wr = __shfl(w, r + 8 * lh, 32);
            float s0 = S0[r], s1 = S1[r];
            s0 = s0 >= 0.f ? s0 : 0.1f * s0;            // LeakyReLU(0.1)
            s1 = s1 >= 0.f ? s1 : 0.1f * s1;
            s0 = ((wr >> l16) & 1u)        ? s0 : 0.f;  // adjacency mask
            s1 = ((wr >> (16 + l16)) & 1u) ? s1 : 0.f;
            if (s0 == 0.f) s0 = -1e16f;                 // masked_fill(att==0)
            if (s1 == 0.f) s1 = -1e16f;

            // row max over the 16-lane half (keys j..j+31)
            float t = fmaxf(s0, s1);
            t = fmaxf(t, __shfl_xor(t, 1));
            t = fmaxf(t, __shfl_xor(t, 2));
            t = fmaxf(t, __shfl_xor(t, 4));
            t = fmaxf(t, __shfl_xor(t, 8));
            float mnew  = fmaxf(mrun[r], t);
            float scale = __expf(mrun[r] - mnew);
            mrun[r] = mnew;

            float e0 = __expf(s0 - mnew);
            float e1 = __expf(s1 - mnew);
            float rs = e0 + e1;
            rs += __shfl_xor(rs, 1);
            rs += __shfl_xor(rs, 2);
            rs += __shfl_xor(rs, 4);
            rs += __shfl_xor(rs, 8);
            lrun[r] = lrun[r] * scale + rs;

            O0[r] *= scale;   // same (reg, lane-half) -> row mapping as S
            O1[r] *= scale;
            p0[r] = e0;
            p1[r] = e1;
        }

        // ---- C-layout -> A-layout for P via per-wave LDS round trip ----
#pragma unroll
        for (int r = 0; r < 8; ++r) {
            int M = r + 8 * lh;
            ps[M * 32 + l16]      = (_Float16)p0[r];
            ps[M * 32 + 16 + l16] = (_Float16)p1[r];
        }
        V16H pa, zb0, zb1;
#pragma unroll
        for (int v = 0; v < 8; ++v) {
            int Kp = 16 * lh + 2 * v;
            pa.u[v]  = *(const unsigned int*)&ps[l16 * 32 + Kp];
            zb0.u[v] = *(const unsigned int*)&zkeyT[ l16       * 32 + 16 * lh + 2 * v];
            zb1.u[v] = *(const unsigned int*)&zkeyT[(16 + l16) * 32 + 16 * lh + 2 * v];
        }
        // ---- O += P (16x32) x Zk (32x32), split into two N=16 tiles ----
        O0 = __builtin_amdgcn_wmma_f32_16x16x32_f16(false, pa.h, false, zb0.h,
                                                    (short)0, O0, false, false);
        O1 = __builtin_amdgcn_wmma_f32_16x16x32_f16(false, pa.h, false, zb1.h,
                                                    (short)0, O1, false, false);
    }

    // ---- normalize and write out ----
    float* ob = out + ((size_t)b * NODES + rowbase) * OUT_DIM;
#pragma unroll
    for (int r = 0; r < 8; ++r) {
        int M = r + 8 * lh;
        float inv = 1.0f / lrun[r];
        ob[M * OUT_DIM + l16]      = O0[r] * inv;
        ob[M * OUT_DIM + 16 + l16] = O1[r] * inv;
    }
}

// ---------------- launcher ----------------
extern "C" void kernel_launch(void* const* d_in, const int* in_sizes, int n_in,
                              void* d_out, int out_size, void* d_ws, size_t ws_size,
                              hipStream_t stream) {
    const float* h  = (const float*)d_in[0];
    const float* W  = (const float*)d_in[1];
    const int* arow = (const int*)d_in[2];
    const int* acol = (const int*)d_in[3];
    float* out = (float*)d_out;

    // workspace layout: [z_f16: 1 MB][mask bits: 2 MB]
    _Float16* z = (_Float16*)d_ws;
    unsigned int* mask =
        (unsigned int*)((char*)d_ws + (size_t)BC * NODES * OUT_DIM * sizeof(_Float16));

    int nwords = NODES * MWORDS;
    zero_mask_kernel<<<(nwords + 255) / 256, 256, 0, stream>>>(mask, nwords);

    int nedges = in_sizes[2];
    build_mask_kernel<<<(nedges + 255) / 256, 256, 0, stream>>>(arow, acol, mask, nedges);

    int nz = BC * NODES * OUT_DIM;
    compute_z_kernel<<<(nz + 255) / 256, 256, 0, stream>>>(h, W, z);

    gat_attn_kernel<<<BC * (NODES / ROWS_PER_WG), WAVES * 32, 0, stream>>>(z, mask, out);
}
